// dggcn_24696061952066
// MI455X (gfx1250) — matrix operand order repaired and checked by reading, not compiled, not run on hardware
//
#include <hip/hip_runtime.h>
#include <hip/hip_bf16.h>
#include <stddef.h>

// ---------------- problem constants ----------------
constexpr int NB   = 64;    // batch
constexpr int CIN  = 128;
constexpr int COUT = 256;
constexpr int KK   = 3;     // kernel count
constexpr int MID  = 64;    // COUT/4
constexpr int KM   = 192;   // KK*MID
constexpr int KMP  = 96;    // KM/2 channel pairs
constexpr int T    = 128;
constexpr int V    = 25;
constexpr int NTV  = T * V; // 3200 columns per sample
constexpr int TP   = 32;    // padded V stride for pre_x
constexpr float EPS = 1e-5f;

// CDNA5 async global->LDS staging (guarded; manual fallback otherwise)
#if defined(__has_builtin)
#if __has_builtin(__builtin_amdgcn_global_load_async_to_lds_b128) && \
    __has_builtin(__builtin_amdgcn_s_wait_asynccnt)
#define USE_ASYNC_LDS 1
#endif
#endif

// ---------------- WMMA types ----------------
typedef __attribute__((ext_vector_type(16))) __bf16        v16bf;
typedef __attribute__((ext_vector_type(8)))  float         v8f;
typedef __attribute__((ext_vector_type(4)))  unsigned int  v4u;
typedef __attribute__((ext_vector_type(4)))  int           v4i;

#ifdef USE_ASYNC_LDS
typedef __attribute__((address_space(1))) v4i* gv4i_p;   // global v4i*
typedef __attribute__((address_space(3))) v4i* lv4i_p;   // LDS v4i*
#endif

union Frag { v16bf v; v4u q[2]; unsigned int u[8]; };

__device__ __forceinline__ unsigned short f2bfu(float f) {
  union { float f; unsigned u; } a; a.f = f;
  unsigned r = a.u + 0x7FFFu + ((a.u >> 16) & 1u);   // round-to-nearest-even
  return (unsigned short)(r >> 16);
}
__device__ __forceinline__ unsigned int packbf2(float lo, float hi) {
  return (unsigned)f2bfu(lo) | ((unsigned)f2bfu(hi) << 16);
}

// ============================================================
// K0: repack an (Mt*16 x Kdim) f32 weight matrix into bf16
// A-fragment order: dst[((mt*Ks+ks)*32 + lane)*8 + j] = pack(k, k+1)
// ============================================================
__global__ void repack_kernel(const float* __restrict__ W, unsigned int* __restrict__ dst,
                              int Ks, int Kdim, int total) {
  int idx = blockIdx.x * 256 + threadIdx.x;
  if (idx >= total) return;
  int j    = idx & 7;
  int lane = (idx >> 3) & 31;
  int tile = idx >> 8;
  int ks   = tile % Ks, mt = tile / Ks;
  int m = lane & 15, h = lane >> 4;
  int kk = (h << 4) + (j << 1);
  const float* w = W + (size_t)(mt * 16 + m) * Kdim + ks * 32 + kk;
  dst[idx] = packbf2(w[0], w[1]);
}

// ============================================================
// K1: tmp[n,c,v] = mean_t x[n,c,t,v]
// ============================================================
__global__ void mean_kernel(const float* __restrict__ x, float* __restrict__ tmp) {
  int nc = blockIdx.x;              // n*CIN + c
  int lane = threadIdx.x;           // 32 threads
  if (lane >= V) return;
  const float* src = x + (size_t)nc * NTV + lane;
  float s = 0.f;
  #pragma unroll 8
  for (int t = 0; t < T; ++t) s += src[t * V];
  tmp[(size_t)nc * V + lane] = s * (1.f / (float)T);
}

// ============================================================
// K2: x1[n,km,v], x2[n,km,v] = W1/W2 . tmp + b   (tiny GEMMs)
// ============================================================
__global__ void x12_kernel(const float* __restrict__ tmp,
                           const float* __restrict__ W1, const float* __restrict__ b1,
                           const float* __restrict__ W2, const float* __restrict__ b2,
                           float* __restrict__ x1, float* __restrict__ x2) {
  int km = blockIdx.x, n = blockIdx.y;
  int tid = threadIdx.x;            // 64 threads, 50 active
  if (tid >= 2 * V) return;
  int which = tid / V, v = tid % V;
  const float* W  = which ? W2 : W1;
  const float* bb = which ? b2 : b1;
  const float* tn = tmp + (size_t)n * CIN * V + v;
  const float* wr = W + (size_t)km * CIN;
  float s = bb[km];
  #pragma unroll 8
  for (int c = 0; c < CIN; ++c) s += wr[c] * tn[c * V];
  (which ? x2 : x1)[(size_t)(n * KM + km) * V + v] = s;
}

// ============================================================
// K3: A6eff[n,k,c,v,w] = tanh(x1[c,v]-x2[c,w])*alpha0 + A[k,v,w]
//                        + softmax_v(x1^T x2)[v,w]*beta0
// ============================================================
__global__ void adj_kernel(const float* __restrict__ x1, const float* __restrict__ x2,
                           const float* __restrict__ A, const float* __restrict__ alpha,
                           const float* __restrict__ beta, float* __restrict__ A6) {
  __shared__ float s1[MID][V];
  __shared__ float s2[MID][V];
  __shared__ float sg[V][V];
  int nk = blockIdx.x;              // n*KK + k
  int k  = nk % KK;
  int tid = threadIdx.x;            // 256
  const float* b1p = x1 + (size_t)nk * MID * V;
  const float* b2p = x2 + (size_t)nk * MID * V;
  for (int idx = tid; idx < MID * V; idx += 256) {
    s1[idx / V][idx % V] = b1p[idx];
    s2[idx / V][idx % V] = b2p[idx];
  }
  __syncthreads();
  for (int idx = tid; idx < V * V; idx += 256) {
    int v = idx / V, w = idx % V;
    float s = 0.f;
    #pragma unroll 8
    for (int c = 0; c < MID; ++c) s += s1[c][v] * s2[c][w];
    sg[v][w] = s;
  }
  __syncthreads();
  if (tid < V) {                    // column softmax over rows v
    int w = tid;
    float mx = -1e30f;
    for (int v = 0; v < V; ++v) mx = fmaxf(mx, sg[v][w]);
    float sum = 0.f;
    for (int v = 0; v < V; ++v) { float e = __expf(sg[v][w] - mx); sg[v][w] = e; sum += e; }
    float inv = 1.f / sum;
    for (int v = 0; v < V; ++v) sg[v][w] *= inv;
  }
  __syncthreads();
  float a0 = alpha[0], be0 = beta[0];
  const float* Ak = A + (size_t)k * V * V;
  float* outp = A6 + (size_t)nk * MID * V * V;
  for (int idx = tid; idx < MID * V * V; idx += 256) {
    int c = idx / (V * V), r = idx % (V * V);
    int v = r / V, w = r % V;
    outp[idx] = tanhf(s1[c][v] - s2[c][w]) * a0 + Ak[r] + sg[v][w] * be0;
  }
}

// ============================================================
// K4: pre_x(bf16, padded t-stride 32) = relu(bn(W_pre . x))
// GEMM M=192, N=3200 per n, K=128; 4 waves/block, one M-tile each.
// ============================================================
__global__ void pre_kernel(const float* __restrict__ x, const unsigned int* __restrict__ Wp,
                           const float* __restrict__ bpre, const float* __restrict__ gpre,
                           const float* __restrict__ bepre, const float* __restrict__ mpre,
                           const float* __restrict__ vpre,
                           unsigned short* __restrict__ preX) {
  __shared__ alignas(16) unsigned int sB[16 * 64];   // [col][kpair], kpair=c/2
  const int colBase = blockIdx.x * 16;
  const int mTile   = blockIdx.y * 4 + (threadIdx.x >> 5);
  const int n       = blockIdx.z;
  const int tid     = threadIdx.x;       // 0..127
  const float* xn = x + (size_t)n * CIN * NTV;
  for (int idx = tid; idx < 16 * 64; idx += 128) {
    int kp = idx >> 4, col = idx & 15;
    const float* s = xn + (size_t)(2 * kp) * NTV + colBase + col;
    sB[col * 64 + kp] = packbf2(s[0], s[NTV]);
  }
  __syncthreads();
  const int lane = tid & 31;
  const int col  = lane & 15;
  const int h    = lane >> 4;
  v8f acc = {};
  #pragma unroll
  for (int ks = 0; ks < 4; ++ks) {       // K = 128 -> 4 steps of 32
    Frag a, b;
    const unsigned int* ap = Wp + (size_t)(((mTile * 4 + ks) * 32) + lane) * 8;
    a.q[0] = *(const v4u*)ap;
    a.q[1] = *(const v4u*)(ap + 4);
    const unsigned int* bp = &sB[col * 64 + ks * 16 + h * 8];
    b.q[0] = *(const v4u*)bp;
    b.q[1] = *(const v4u*)(bp + 4);
    acc = __builtin_amdgcn_wmma_f32_16x16x32_bf16(false, a.v, false, b.v, (short)0, acc, false, false);
  }
  #pragma unroll
  for (int r = 0; r < 8; ++r) {          // folded BN + ReLU -> bf16 (padded layout)
    int km = mTile * 16 + r + h * 8;
    float s  = gpre[km] * rsqrtf(vpre[km] + EPS);
    float bb = (bpre[km] - mpre[km]) * s + bepre[km];
    float val = acc[r] * s + bb;
    val = val > 0.f ? val : 0.f;
    int tv = colBase + col;
    int t = tv / V, v = tv - t * V;
    preX[((size_t)(n * KM + km) * T + t) * TP + v] = f2bfu(val);
  }
}

// ============================================================
// K5: spatial mix for a CHANNEL PAIR (kc0=2kp, kc0+1):
// y dword yP[(n*96+kp)*NTV + t*25+w] = pack(bf16(y_kc0), bf16(y_kc1))
// -> out_kernel's B tile becomes a pure dword copy (async-friendly).
// ============================================================
__global__ void spatial_kernel(const unsigned short* __restrict__ preX,
                               const float* __restrict__ A6,
                               unsigned int* __restrict__ yP) {
  __shared__ alignas(16) unsigned int sBp[2][32 * 16];  // per channel: [col(w)][vpair]
  const int nkp = blockIdx.x;            // n*96 + kp
  const int n   = nkp / KMP, kp = nkp - n * KMP;
  const int kc0 = 2 * kp;
  const int tid = threadIdx.x;           // 256 = 8 waves
  const float* a6 = A6 + ((size_t)n * KM + kc0) * (V * V);
  if (tid == 0) __builtin_prefetch(a6 + 2 * V * V, 0, 1);   // next block's A6
  for (int idx = tid; idx < 2 * 512; idx += 256) {
    int sub = idx >> 9, r = idx & 511;
    int col = r & 31, kq = r >> 5;       // kq = v/2 pair
    const float* s = a6 + (size_t)sub * (V * V);
    int v0 = 2 * kq;
    float lo = (v0     < V && col < V) ? s[v0 * V + col] : 0.f;
    float hi = (v0 + 1 < V && col < V) ? s[(v0 + 1) * V + col] : 0.f;
    sBp[sub][col * 16 + kq] = packbf2(lo, hi);
  }
  __syncthreads();
  const int wave = tid >> 5;             // t-tile 0..7
  const int lane = tid & 31;
  const int col  = lane & 15, h = lane >> 4;
  const unsigned short* ar0 = preX + (((size_t)n * KM + kc0) * T + wave * 16 + col) * TP;
  const unsigned short* ar1 = ar0 + (size_t)T * TP;
  Frag a0, a1;
  a0.q[0] = *(const v4u*)(ar0 + h * 8);        // K = h*8..h*8+7 (valid)
  a0.q[1] = *(const v4u*)(ar0 + 16 + h * 8);   // K = 16+h*8..  (h=1: only K=24 valid)
  a1.q[0] = *(const v4u*)(ar1 + h * 8);
  a1.q[1] = *(const v4u*)(ar1 + 16 + h * 8);
  if (h) {                                     // mask pad garbage K>=25
    a0.u[4] &= 0xFFFFu; a0.u[5] = 0u; a0.u[6] = 0u; a0.u[7] = 0u;
    a1.u[4] &= 0xFFFFu; a1.u[5] = 0u; a1.u[6] = 0u; a1.u[7] = 0u;
  }
  unsigned int* ybase = yP + (size_t)nkp * NTV;
  #pragma unroll
  for (int ntile = 0; ntile < 2; ++ntile) {
    Frag b0, b1;
    const unsigned int* bp0 = &sBp[0][(ntile * 16 + col) * 16];
    const unsigned int* bp1 = &sBp[1][(ntile * 16 + col) * 16];
    b0.q[0] = *(const v4u*)(bp0 + h * 4); b0.q[1] = *(const v4u*)(bp0 + 8 + h * 4);
    b1.q[0] = *(const v4u*)(bp1 + h * 4); b1.q[1] = *(const v4u*)(bp1 + 8 + h * 4);
    v8f acc0 = {}, acc1 = {};
    acc0 = __builtin_amdgcn_wmma_f32_16x16x32_bf16(false, a0.v, false, b0.v, (short)0, acc0, false, false);
    acc1 = __builtin_amdgcn_wmma_f32_16x16x32_bf16(false, a1.v, false, b1.v, (short)0, acc1, false, false);
    int w = ntile * 16 + col;
    if (w < V) {
      #pragma unroll
      for (int r = 0; r < 8; ++r) {
        int t = wave * 16 + r + h * 8;
        ybase[t * V + w] = packbf2(acc0[r], acc1[r]);   // lo=kc0, hi=kc1
      }
    }
  }
}

// ============================================================
// K6: out = relu( bn(W_post . y) + bn(W_down . x) )   fused two GEMMs
// y tile staged by pure dword copy (async global->LDS when available).
// ============================================================
__global__ void out_kernel(const float* __restrict__ x, const unsigned int* __restrict__ yP,
                           const unsigned int* __restrict__ WpostP,
                           const unsigned int* __restrict__ WdownP,
                           const float* __restrict__ bpost,
                           const float* __restrict__ bdown,
                           const float* __restrict__ gdn, const float* __restrict__ bedn,
                           const float* __restrict__ mdn, const float* __restrict__ vdn,
                           const float* __restrict__ gbn, const float* __restrict__ bebn,
                           const float* __restrict__ mbn, const float* __restrict__ vbn,
                           float* __restrict__ out) {
  __shared__ alignas(16) unsigned int sYp[KMP * 16];  // [kp][col]  row-major (copy image)
  __shared__ alignas(16) unsigned int sXp[16 * 64];   // [col][kpair]
  const int colBase = blockIdx.x * 16;
  const int mTile   = blockIdx.y * 8 + (threadIdx.x >> 5);
  const int n       = blockIdx.z;
  const int tid     = threadIdx.x;       // 256
  const unsigned int* yPn = yP + (size_t)n * KMP * NTV;
  const float* xn = x + (size_t)n * CIN * NTV;
  if (tid < KMP)                          // warm L2 for the next column block
    __builtin_prefetch(yPn + (size_t)tid * NTV + colBase + 16, 0, 1);
#ifdef USE_ASYNC_LDS
  for (int q = tid; q < KMP * 4; q += 256) {        // 16B per lane, async
    int kp = q >> 2, c4 = (q & 3) * 4;
    const unsigned int* g = yPn + (size_t)kp * NTV + colBase + c4;
    __builtin_amdgcn_global_load_async_to_lds_b128(
        (gv4i_p)(void*)g, (lv4i_p)(void*)&sYp[kp * 16 + c4], 0, 0);
  }
#else
  for (int idx = tid; idx < KMP * 16; idx += 256) {
    int kp = idx >> 4, c = idx & 15;
    sYp[idx] = yPn[(size_t)kp * NTV + colBase + c];
  }
#endif
  for (int idx = tid; idx < 16 * 64; idx += 256) {
    int kp = idx >> 4, c = idx & 15;
    const float* s = xn + (size_t)(2 * kp) * NTV + colBase + c;
    sXp[c * 64 + kp] = packbf2(s[0], s[NTV]);
  }
#ifdef USE_ASYNC_LDS
  __builtin_amdgcn_s_wait_asynccnt(0);
#endif
  __syncthreads();
  const int lane = tid & 31;
  const int col  = lane & 15, h = lane >> 4;
  v8f acc1 = {}, acc2 = {};
  #pragma unroll
  for (int ks = 0; ks < 6; ++ks) {       // W_post: K=192
    Frag a, b;
    const unsigned int* ap = WpostP + (size_t)(((mTile * 6 + ks) * 32) + lane) * 8;
    a.q[0] = *(const v4u*)ap;
    a.q[1] = *(const v4u*)(ap + 4);
    #pragma unroll
    for (int j = 0; j < 8; ++j) b.u[j] = sYp[(ks * 16 + h * 8 + j) * 16 + col];
    acc1 = __builtin_amdgcn_wmma_f32_16x16x32_bf16(false, a.v, false, b.v, (short)0, acc1, false, false);
  }
  #pragma unroll
  for (int ks = 0; ks < 4; ++ks) {       // W_down: K=128
    Frag a, b;
    const unsigned int* ap = WdownP + (size_t)(((mTile * 4 + ks) * 32) + lane) * 8;
    a.q[0] = *(const v4u*)ap;
    a.q[1] = *(const v4u*)(ap + 4);
    const unsigned int* bp = &sXp[col * 64 + ks * 16 + h * 8];
    b.q[0] = *(const v4u*)bp;
    b.q[1] = *(const v4u*)(bp + 4);
    acc2 = __builtin_amdgcn_wmma_f32_16x16x32_bf16(false, a.v, false, b.v, (short)0, acc2, false, false);
  }
  #pragma unroll
  for (int r = 0; r < 8; ++r) {
    int o = mTile * 16 + r + h * 8;
    float s1 = gbn[o] * rsqrtf(vbn[o] + EPS);
    float c1 = (bpost[o] - mbn[o]) * s1 + bebn[o];
    float s2 = gdn[o] * rsqrtf(vdn[o] + EPS);
    float c2 = (bdown[o] - mdn[o]) * s2 + bedn[o];
    float val = acc1[r] * s1 + acc2[r] * s2 + c1 + c2;
    out[(size_t)(n * COUT + o) * NTV + colBase + col] = val > 0.f ? val : 0.f;
  }
}

// ============================================================
extern "C" void kernel_launch(void* const* d_in, const int* in_sizes, int n_in,
                              void* d_out, int out_size, void* d_ws, size_t ws_size,
                              hipStream_t stream) {
  const float* x      = (const float*)d_in[0];
  const float* A      = (const float*)d_in[1];
  const float* alpha  = (const float*)d_in[2];
  const float* beta   = (const float*)d_in[3];
  const float* W_pre  = (const float*)d_in[4];
  const float* b_pre  = (const float*)d_in[5];
  const float* g_pre  = (const float*)d_in[6];
  const float* be_pre = (const float*)d_in[7];
  const float* m_pre  = (const float*)d_in[8];
  const float* v_pre  = (const float*)d_in[9];
  const float* W1     = (const float*)d_in[10];
  const float* b1     = (const float*)d_in[11];
  const float* W2     = (const float*)d_in[12];
  const float* b2     = (const float*)d_in[13];
  const float* W_post = (const float*)d_in[14];
  const float* b_post = (const float*)d_in[15];
  const float* W_down = (const float*)d_in[16];
  const float* b_down = (const float*)d_in[17];
  const float* g_down = (const float*)d_in[18];
  const float* be_down= (const float*)d_in[19];
  const float* m_down = (const float*)d_in[20];
  const float* v_down = (const float*)d_in[21];
  const float* g_bn   = (const float*)d_in[22];
  const float* be_bn  = (const float*)d_in[23];
  const float* m_bn   = (const float*)d_in[24];
  const float* v_bn   = (const float*)d_in[25];
  float* out = (float*)d_out;

  // workspace layout (256B aligned regions)
  char* ws = (char*)d_ws;
  size_t off = 0;
  auto take = [&](size_t bytes) { void* p = ws + off; off += (bytes + 255) & ~(size_t)255; return p; };
  float* tmp = (float*)take((size_t)NB * CIN * V * 4);            // 0.82 MB
  float* x1  = (float*)take((size_t)NB * KM * V * 4);             // 1.23 MB
  float* x2  = (float*)take((size_t)NB * KM * V * 4);             // 1.23 MB
  float* A6  = (float*)take((size_t)NB * KM * V * V * 4);         // 30.7 MB
  unsigned short* preX = (unsigned short*)take((size_t)NB * KM * T * TP * 2); // 100.7 MB (padded)
  unsigned int*   yP   = (unsigned int*)take((size_t)NB * KMP * NTV * 4);     // 78.6 MB (pair-packed)
  unsigned int* WpreP  = (unsigned int*)take((size_t)12 * 4 * 256 * 4);       // 48 KB
  unsigned int* WpostP = (unsigned int*)take((size_t)16 * 6 * 256 * 4);       // 96 KB
  unsigned int* WdownP = (unsigned int*)take((size_t)16 * 4 * 256 * 4);       // 64 KB

  // weight repack (fragment-ready bf16)
  repack_kernel<<<dim3((12 * 4 * 256) / 256), dim3(256), 0, stream>>>(W_pre,  WpreP,  4, CIN, 12 * 4 * 256);
  repack_kernel<<<dim3((16 * 6 * 256) / 256), dim3(256), 0, stream>>>(W_post, WpostP, 6, KM,  16 * 6 * 256);
  repack_kernel<<<dim3((16 * 4 * 256) / 256), dim3(256), 0, stream>>>(W_down, WdownP, 4, CIN, 16 * 4 * 256);

  mean_kernel<<<dim3(NB * CIN), dim3(32), 0, stream>>>(x, tmp);
  x12_kernel<<<dim3(KM, NB), dim3(64), 0, stream>>>(tmp, W1, b1, W2, b2, x1, x2);
  adj_kernel<<<dim3(NB * KK), dim3(256), 0, stream>>>(x1, x2, A, alpha, beta, A6);
  pre_kernel<<<dim3(NTV / 16, KM / 64, NB), dim3(128), 0, stream>>>(
      x, WpreP, b_pre, g_pre, be_pre, m_pre, v_pre, preX);
  spatial_kernel<<<dim3(NB * KMP), dim3(256), 0, stream>>>(preX, A6, yP);
  out_kernel<<<dim3(NTV / 16, COUT / 128, NB), dim3(256), 0, stream>>>(
      x, yP, WpostP, WdownP, b_post, b_down,
      g_down, be_down, m_down, v_down, g_bn, be_bn, m_bn, v_bn, out);
}
